// SelfPointBranch_32392643346946
// MI455X (gfx1250) — compile-verified, matrix operand order checked by reading, not run on hardware
//
#include <hip/hip_runtime.h>
#include <hip/hip_bf16.h>

typedef _Float16 half8 __attribute__((ext_vector_type(8)));
typedef _Float16 v16h  __attribute__((ext_vector_type(16)));
typedef float    v8f   __attribute__((ext_vector_type(8)));

#define BB 4
#define CC 256
#define CQ 32
#define NN 4096

// ---- WMMA helpers -----------------------------------------------------------

__device__ __forceinline__ v8f wmma_f16(v16h a, v16h b, v8f c) {
  // D = A(16x32 f16) * B(32x16 f16) + C(16x16 f32)
  return __builtin_amdgcn_wmma_f32_16x16x32_f16(false, a, false, b, (short)0, c,
                                                false, false);
}

// Load a 16x32 f16 operand tile (A: rows x K, or B stored as [col][K]).
// Memory is row-major with leading dimension `ld` (in halves).
// Lane l%16 owns row/col (l%16); K chunks split per ISA layout:
//   VGPR0-3: K = hi*8 .. hi*8+7 ; VGPR4-7: K = 16+hi*8 .. 16+hi*8+7
__device__ __forceinline__ v16h load_tile16x32(const _Float16* __restrict__ p,
                                               int ld, int lane) {
  int r  = lane & 15;
  int hi = (lane >> 4) & 1;
  const half8* p0 = reinterpret_cast<const half8*>(p + (size_t)r * ld + hi * 8);
  const half8* p1 = reinterpret_cast<const half8*>(p + (size_t)r * ld + 16 + hi * 8);
  half8 a = *p0;
  half8 b = *p1;
  return __builtin_shufflevector(a, b, 0, 1, 2, 3, 4, 5, 6, 7,
                                 8, 9, 10, 11, 12, 13, 14, 15);
}

// Async-copy a contiguous 2KB K-chunk (32 rows x 32 halves) into LDS.
// Issued by one wave: 4 x b128 per lane. Tracked by ASYNCcnt.
__device__ __forceinline__ void async_copy_2k(const _Float16* __restrict__ src,
                                              _Float16* dst_lds, int lane) {
  unsigned lds = (unsigned)(uintptr_t)dst_lds;  // low 32 bits = LDS byte offset
  const char* g = (const char*)src;
#pragma unroll
  for (int j = 0; j < 4; ++j) {
    unsigned la = lds + (unsigned)(lane * 16 + j * 512);
    const char* ga = g + lane * 16 + j * 512;
    asm volatile("global_load_async_to_lds_b128 %0, %1, off"
                 :: "v"(la), "v"(ga) : "memory");
  }
}

#define WAIT_ASYNC0() asm volatile("s_wait_asynccnt 0x0" ::: "memory")
#define WAIT_DS0()    asm volatile("s_wait_dscnt 0x0" ::: "memory")

// ---- Kernel 1: x[b][c][n] f32 -> xh[b][n][c] f16 (LDS tile transpose) -------

__global__ void k_transpose_cast(const float* __restrict__ x,
                                 _Float16* __restrict__ xh) {
  __shared__ float tile[32][33];
  int t  = threadIdx.x;
  int tx = t & 31;
  int ty = t >> 5;                 // 0..7
  int blk   = blockIdx.x;
  int ntile = blk & 127;           // N/32
  int ctile = (blk >> 7) & 7;      // C/32
  int b     = blk >> 10;

  const float* xp = x + ((size_t)b * CC + (size_t)ctile * 32) * NN + (size_t)ntile * 32;
#pragma unroll
  for (int i = 0; i < 4; ++i) {
    int c = ty + i * 8;
    tile[c][tx] = xp[(size_t)c * NN + tx];
  }
  __syncthreads();
  _Float16* op = xh + ((size_t)b * NN + (size_t)ntile * 32) * CC + (size_t)ctile * 32;
#pragma unroll
  for (int i = 0; i < 4; ++i) {
    int n = ty + i * 8;
    op[(size_t)n * CC + tx] = (_Float16)tile[tx][n];
  }
}

// ---- Kernel 2: f32 -> f16 elementwise (weights) -----------------------------

__global__ void k_cast_f16(const float* __restrict__ in,
                           _Float16* __restrict__ out, int n) {
  int i = blockIdx.x * blockDim.x + threadIdx.x;
  if (i < n) out[i] = (_Float16)in[i];
}

// ---- Kernel 3: projections q,k,v via WMMA -----------------------------------
// One wave = one (b, n-tile, 16-wide output column group).
// groups: 0-1 -> q, 2-3 -> k, 4-19 -> v.

__global__ void __launch_bounds__(128) k_proj(
    const _Float16* __restrict__ xh,
    const _Float16* __restrict__ wqh, const float* __restrict__ bq,
    const _Float16* __restrict__ wkh, const float* __restrict__ bk,
    const _Float16* __restrict__ wvh, const float* __restrict__ bv,
    _Float16* __restrict__ qh, _Float16* __restrict__ kh,
    _Float16* __restrict__ vh) {
  int lane = threadIdx.x & 31;
  int wid  = threadIdx.x >> 5;
  int w    = blockIdx.x * 4 + wid;
  int group = w % 20;
  int ntile = (w / 20) % (NN / 16);
  int b     = w / (20 * (NN / 16));
  int nbase = ntile * 16;

  const _Float16* W;
  const float* bias;
  int obase, kind;
  if (group < 2)      { kind = 0; W = wqh; bias = bq; obase = group * 16; }
  else if (group < 4) { kind = 1; W = wkh; bias = bk; obase = (group - 2) * 16; }
  else                { kind = 2; W = wvh; bias = bv; obase = (group - 4) * 16; }

  const _Float16* xp = xh + ((size_t)b * NN + nbase) * CC;
  v8f acc = {};
#pragma unroll
  for (int kc = 0; kc < CC; kc += 32) {
    v16h a  = load_tile16x32(xp + kc, CC, lane);
    v16h bm = load_tile16x32(W + (size_t)obase * CC + kc, CC, lane);
    acc = wmma_f16(a, bm, acc);
  }

  int l  = lane & 15;
  int hi = lane >> 4;
  float bb = bias[obase + l];
#pragma unroll
  for (int r = 0; r < 8; ++r) acc[r] += bb;

  if (kind < 2) {
    // q/k stored [b][n][o] f16 (operand layout for the S GEMM)
    _Float16* dst = (kind ? kh : qh) + ((size_t)b * NN + nbase) * CQ;
#pragma unroll
    for (int r = 0; r < 8; ++r)
      dst[(size_t)(r + 8 * hi) * CQ + obase + l] = (_Float16)acc[r];
  } else {
    // v stored [b][c][m] f16 : lane's 8 rows are contiguous m -> one b128 store
    half8 pk;
#pragma unroll
    for (int r = 0; r < 8; ++r) pk[r] = (_Float16)acc[r];
    *reinterpret_cast<half8*>(
        vh + ((size_t)b * CC + obase + l) * NN + nbase + 8 * hi) = pk;
  }
}

// ---- Kernel 4: flash attention + epilogue -----------------------------------
// One wave = (b, 16-row q-tile, half of the 256 channels).
// K chunks (2KB, identical for all 4 waves of a block) are staged into LDS by
// wave 0 via GLOBAL_LOAD_ASYNC_TO_LDS_B128, double-buffered, with
// s_wait_asynccnt completion and barrier handoff. V is read directly (L2-hot).
// Pass 1: online row max & exp-sum. Pass 2: recompute S, P=exp(S-m)/l staged
// through LDS (D-layout -> A-layout), accumulate O = P*V^T, fuse gamma*O + x.

__global__ void __launch_bounds__(128) k_attn(
    const _Float16* __restrict__ qh, const _Float16* __restrict__ kh,
    const _Float16* __restrict__ vh, const float* __restrict__ x,
    const float* __restrict__ gamma, float* __restrict__ out) {
  __shared__ __align__(16) _Float16 kbuf[2][32 * 32];   // double-buffered K chunk
  __shared__ __align__(16) _Float16 plds[4][16 * 40];   // per-wave P staging
  int lane = threadIdx.x & 31;
  int wid  = threadIdx.x >> 5;
  int w     = blockIdx.x * 4 + wid;
  int chalf = w & 1;
  int ntile = (w >> 1) & 255;
  int b     = w >> 9;
  int nbase = ntile * 16;
  int l  = lane & 15;
  int hi = lane >> 4;
  const int NCH = NN / 32;

  v16h aq = load_tile16x32(qh + ((size_t)b * NN + nbase) * CQ, CQ, lane);
  const _Float16* kb = kh + (size_t)b * NN * CQ;
  v8f zero = {};

  // ---- pass 1: softmax stats (row m = r + 8*hi per lane) ----
  float mx[8], sm[8];
#pragma unroll
  for (int r = 0; r < 8; ++r) { mx[r] = -3.0e38f; sm[r] = 0.0f; }

  if (wid == 0) async_copy_2k(kb, &kbuf[0][0], lane);
  for (int it = 0; it < NCH; ++it) {
    int cur = it & 1;
    if (wid == 0) WAIT_ASYNC0();
    __syncthreads();                                   // K chunk `it` ready
    v16h b0 = load_tile16x32(&kbuf[cur][0], 32, lane);
    v16h b1 = load_tile16x32(&kbuf[cur][16 * 32], 32, lane);
    WAIT_DS0();
    __syncthreads();                                   // all waves done reading
    if (wid == 0 && it + 1 < NCH)
      async_copy_2k(kb + (size_t)(it + 1) * 32 * CQ, &kbuf[cur ^ 1][0], lane);
    v8f s0 = wmma_f16(aq, b0, zero);
    v8f s1 = wmma_f16(aq, b1, zero);
#pragma unroll
    for (int r = 0; r < 8; ++r) {
      float t = fmaxf(s0[r], s1[r]);
      t = fmaxf(t, __shfl_xor(t, 1, 32));
      t = fmaxf(t, __shfl_xor(t, 2, 32));
      t = fmaxf(t, __shfl_xor(t, 4, 32));
      t = fmaxf(t, __shfl_xor(t, 8, 32));
      float nm = fmaxf(mx[r], t);
      float e = __expf(s0[r] - nm) + __expf(s1[r] - nm);
      e += __shfl_xor(e, 1, 32);
      e += __shfl_xor(e, 2, 32);
      e += __shfl_xor(e, 4, 32);
      e += __shfl_xor(e, 8, 32);
      sm[r] = sm[r] * __expf(mx[r] - nm) + e;
      mx[r] = nm;
    }
  }
  float rl[8];
#pragma unroll
  for (int r = 0; r < 8; ++r) rl[r] = 1.0f / sm[r];

  // ---- pass 2: O += P * V^T ----
  v8f acc[8];
#pragma unroll
  for (int g = 0; g < 8; ++g) acc[g] = zero;

  _Float16* pw = &plds[wid][0];
  const _Float16* vbase = vh + ((size_t)b * CC + (size_t)chalf * 128) * NN;

  if (wid == 0) async_copy_2k(kb, &kbuf[0][0], lane);
  for (int it = 0; it < NCH; ++it) {
    int cur = it & 1;
    int m0  = it * 32;
    if (wid == 0) WAIT_ASYNC0();
    __syncthreads();                                   // K chunk `it` ready
    v16h b0 = load_tile16x32(&kbuf[cur][0], 32, lane);
    v16h b1 = load_tile16x32(&kbuf[cur][16 * 32], 32, lane);
    WAIT_DS0();
    __syncthreads();                                   // all waves done reading
    if (wid == 0 && it + 1 < NCH)
      async_copy_2k(kb + (size_t)(it + 1) * 32 * CQ, &kbuf[cur ^ 1][0], lane);
    v8f s0 = wmma_f16(aq, b0, zero);
    v8f s1 = wmma_f16(aq, b1, zero);
#pragma unroll
    for (int r = 0; r < 8; ++r) {
      float p0 = __expf(s0[r] - mx[r]) * rl[r];
      float p1 = __expf(s1[r] - mx[r]) * rl[r];
      int n = r + 8 * hi;  // row in the 16-row tile
      pw[n * 40 + l]      = (_Float16)p0;   // m-local = l
      pw[n * 40 + 16 + l] = (_Float16)p1;   // m-local = 16+l
    }
    WAIT_DS0();                                        // stores visible in-wave
    v16h pa = load_tile16x32(pw, 40, lane);            // D-layout -> A-layout
#pragma unroll
    for (int g = 0; g < 8; ++g) {
      v16h bv_ = load_tile16x32(vbase + (size_t)g * 16 * NN + m0, NN, lane);
      acc[g] = wmma_f16(pa, bv_, acc[g]);
    }
  }

  // ---- epilogue: out = gamma * O + x (residual), vectorized b128 ----
  float gm = gamma[0];
#pragma unroll
  for (int g = 0; g < 8; ++g) {
    int c = chalf * 128 + g * 16 + l;
    size_t base = ((size_t)b * CC + c) * NN + nbase + 8 * hi;
    const float4* xp = reinterpret_cast<const float4*>(x + base);
    float4 x0 = xp[0];
    float4 x1 = xp[1];
    float4 o0, o1;
    o0.x = gm * acc[g][0] + x0.x;
    o0.y = gm * acc[g][1] + x0.y;
    o0.z = gm * acc[g][2] + x0.z;
    o0.w = gm * acc[g][3] + x0.w;
    o1.x = gm * acc[g][4] + x1.x;
    o1.y = gm * acc[g][5] + x1.y;
    o1.z = gm * acc[g][6] + x1.z;
    o1.w = gm * acc[g][7] + x1.w;
    float4* op = reinterpret_cast<float4*>(out + base);
    op[0] = o0;
    op[1] = o1;
  }
}

// ---- host launcher ----------------------------------------------------------

extern "C" void kernel_launch(void* const* d_in, const int* in_sizes, int n_in,
                              void* d_out, int out_size, void* d_ws, size_t ws_size,
                              hipStream_t stream) {
  (void)in_sizes; (void)n_in; (void)out_size; (void)ws_size;
  const float* x     = (const float*)d_in[0];
  const float* wq    = (const float*)d_in[1];
  const float* bq    = (const float*)d_in[2];
  const float* wk    = (const float*)d_in[3];
  const float* bk    = (const float*)d_in[4];
  const float* wv    = (const float*)d_in[5];
  const float* bv    = (const float*)d_in[6];
  const float* gamma = (const float*)d_in[7];
  float* out = (float*)d_out;

  char* ws = (char*)d_ws;
  size_t off = 0;
  auto carve = [&](size_t bytes) -> void* {
    void* p = ws + off;
    off += (bytes + 255) & ~(size_t)255;
    return p;
  };
  _Float16* xh  = (_Float16*)carve((size_t)BB * NN * CC * sizeof(_Float16));
  _Float16* vh  = (_Float16*)carve((size_t)BB * CC * NN * sizeof(_Float16));
  _Float16* qh  = (_Float16*)carve((size_t)BB * NN * CQ * sizeof(_Float16));
  _Float16* kh  = (_Float16*)carve((size_t)BB * NN * CQ * sizeof(_Float16));
  _Float16* wqh = (_Float16*)carve((size_t)CQ * CC * sizeof(_Float16));
  _Float16* wkh = (_Float16*)carve((size_t)CQ * CC * sizeof(_Float16));
  _Float16* wvh = (_Float16*)carve((size_t)CC * CC * sizeof(_Float16));

  // 1) x -> xh (f16, [b][n][c])
  k_transpose_cast<<<BB * (CC / 32) * (NN / 32), 256, 0, stream>>>(x, xh);

  // 2) weights -> f16
  k_cast_f16<<<(CQ * CC + 255) / 256, 256, 0, stream>>>(wq, wqh, CQ * CC);
  k_cast_f16<<<(CQ * CC + 255) / 256, 256, 0, stream>>>(wk, wkh, CQ * CC);
  k_cast_f16<<<(CC * CC + 255) / 256, 256, 0, stream>>>(wv, wvh, CC * CC);

  // 3) projections: 4 b * 256 n-tiles * 20 groups waves, 4 waves/block
  k_proj<<<(BB * (NN / 16) * 20) / 4, 128, 0, stream>>>(
      xh, wqh, bq, wkh, bk, wvh, bv, qh, kh, vh);

  // 4) attention + residual: 4 b * 256 n-tiles * 2 channel-halves waves
  k_attn<<<(BB * (NN / 16) * 2) / 4, 128, 0, stream>>>(qh, kh, vh, x, gamma, out);
}
